// RTFSBlock_77343771067106
// MI455X (gfx1250) — matrix operand order repaired and checked by reading, not compiled, not run on hardware
//
#include <hip/hip_runtime.h>

#define EPSV 1e-5f

typedef __bf16 bf16;
typedef __attribute__((ext_vector_type(16))) __bf16 v16bf;
typedef __attribute__((ext_vector_type(8)))  float  v8f;

// ---------------- small device helpers ----------------
__device__ __forceinline__ bf16 tobf(float f) {
  union { float f; unsigned u; } v; v.f = f;
  unsigned u = v.u + 0x7fffu + ((v.u >> 16) & 1u);   // round-to-nearest-even
  union { unsigned short s; bf16 b; } o; o.s = (unsigned short)(u >> 16);
  return o.b;
}
__device__ __forceinline__ float wred(float v) {
  #pragma unroll
  for (int m = 16; m; m >>= 1) v += __shfl_xor(v, m, 32);
  return v;
}
__device__ __forceinline__ float wredmax(float v) {
  #pragma unroll
  for (int m = 16; m; m >>= 1) v = fmaxf(v, __shfl_xor(v, m, 32));
  return v;
}
__device__ __forceinline__ float sigm(float x) { return 1.f / (1.f + __expf(-x)); }

// ---------------- problem dims ----------------
#define BB   2
#define CINC 256
#define CHC  64
#define TT0  500
#define FF0  130
#define TF0  65000L
#define TT1  249
#define FF1  64
#define FDD  31
#define TF1  7719L        // TT1*FDD
#define PERB1 494016L     // CHC*TT1*FDD
#define NHH  4
#define QKDD 4
#define CVV  16
#define NW1  24
#define NW2  242
#define NSEQ1 498
#define NSEQ2 62
#define MU1  11952        // NSEQ1*NW1
#define MU2  15004        // NSEQ2*NW2
#define PATT 15438        // BB*TT1*FDD

// =====================================================================
// Generic bf16 WMMA GEMM:  C[b][m*rs + n*cs] = sum_k A[b][m][k]*W[b][n][k] (+bias[n])
// K must be a multiple of 32 (callers pad). One wave per 16x16 tile.
// =====================================================================
__global__ void k_gemm(const bf16* __restrict__ A, long aB, int lda,
                       const bf16* __restrict__ W, long wB, int ldw,
                       float* __restrict__ C, long cB, long rs, long cs,
                       const float* __restrict__ bias,
                       int M, int N, int K) {
  int lane = threadIdx.x & 31;
  int wv   = threadIdx.x >> 5;
  int tilesM = (M + 15) >> 4, tilesN = (N + 15) >> 4;
  int tile = blockIdx.x * 4 + wv;
  if (tile >= tilesM * tilesN) return;          // wave-uniform
  int tm = (tile % tilesM) * 16;
  int tn = (tile / tilesM) * 16;

  const bf16* Ab = A + (long)blockIdx.y * aB;
  const bf16* Wb = W + (long)blockIdx.y * wB;
  float*      Cb = C + (long)blockIdx.y * cB;

  int m = lane & 15, half = lane >> 4;
  int arow = tm + m; if (arow >= M) arow = M - 1;     // clamp reads on edge tiles
  int nrow = tn + m; if (nrow >= N) nrow = N - 1;
  const bf16* ap = Ab + (long)arow * lda;
  const bf16* wp = Wb + (long)nrow * ldw + half * 16;

  v8f acc = {};
  for (int kk = 0; kk < K; kk += 32) {
    v16bf av, bv;
    #pragma unroll
    for (int v = 0; v < 8; ++v) {
      // A k order used here: half 0 -> K 0..15 ; half 1 -> K 16..31 (matches B)
      int k0 = kk + half * 16 + v * 2;
      av[2 * v]     = ap[k0];
      av[2 * v + 1] = ap[k0 + 1];
    }
    #pragma unroll
    for (int e = 0; e < 16; ++e) bv[e] = wp[kk + e];  // B: lane=N, contiguous K per half
    acc = __builtin_amdgcn_wmma_f32_16x16x32_bf16(false, av, false, bv,
                                                  (short)0, acc, false, false);
  }
  int n = tn + (lane & 15);
  if (n >= N) return;
  float bb = bias ? bias[n] : 0.f;
  #pragma unroll
  for (int r = 0; r < 8; ++r) {
    int mr = tm + r + half * 8;
    if (mr < M) Cb[(long)mr * rs + (long)n * cs] = acc[r] + bb;
  }
}

// ---------------- utility kernels ----------------
__global__ void k_cvt(const float* __restrict__ s, bf16* __restrict__ d, long n) {
  long i = (long)blockIdx.x * 256 + threadIdx.x;
  if (i < n) d[i] = tobf(s[i]);
}
__global__ void k_addb(const float* a, const float* b, float* d, int n) {
  int i = blockIdx.x * 256 + threadIdx.x;
  if (i < n) d[i] = a[i] + b[i];
}
__global__ void k_zero(float* p, long n) {
  long i = (long)blockIdx.x * 256 + threadIdx.x;
  if (i < n) p[i] = 0.f;
}
__global__ void k_addres(float* o, const float* x, long n) {
  long i = (long)blockIdx.x * 256 + threadIdx.x;
  if (i < n) o[i] += x[i];
}

// xT[b][tf][ci] = bf16(x[b][ci][tf])
__global__ void k_pack_x(const float* __restrict__ x, bf16* __restrict__ xT) {
  long i = (long)blockIdx.x * 256 + threadIdx.x;
  if (i >= (long)BB * CINC * TF0) return;
  int ci = (int)(i & 255);
  long rest = i >> 8;
  long tf = rest % TF0; int b = (int)(rest / TF0);
  xT[i] = tobf(x[((long)b * CINC + ci) * TF0 + tf]);
}

// h1 = relu(bn(dwconv4x4s2(h0)))
__global__ void k_dw1(const float* __restrict__ h0, float* __restrict__ h1,
                      const float* w, const float* bc, const float* g,
                      const float* beta, const float* mean, const float* var) {
  long i = (long)blockIdx.x * 256 + threadIdx.x;
  if (i >= (long)BB * CHC * TT1 * FF1) return;
  int f = (int)(i % FF1); int t = (int)((i / FF1) % TT1);
  int c = (int)((i / ((long)FF1 * TT1)) & 63); int b = (int)(i / ((long)FF1 * TT1 * CHC));
  const float* src = h0 + ((long)(b * CHC + c)) * TT0 * FF0;
  float s = 0.f;
  #pragma unroll
  for (int ii = 0; ii < 4; ++ii)
    #pragma unroll
    for (int jj = 0; jj < 4; ++jj)
      s += src[(2 * t + ii) * FF0 + 2 * f + jj] * w[c * 16 + ii * 4 + jj];
  s += bc[c];
  float sc = g[c] * rsqrtf(var[c] + EPSV);
  s = s * sc + (beta[c] - mean[c] * sc);
  h1[i] = s > 0.f ? s : 0.f;
}

// x_dp[b][c][t][f<31] = h1[...,f] + adaptive_pool(h0)(t,f)
__global__ void k_pool_add(const float* __restrict__ h0, const float* __restrict__ h1,
                           float* __restrict__ xdp) {
  long i = (long)blockIdx.x * 256 + threadIdx.x;
  if (i >= (long)BB * CHC * TT1 * FDD) return;
  int f = (int)(i % FDD); int t = (int)((i / FDD) % TT1);
  int c = (int)((i / TF1) & 63); int b = (int)(i / PERB1);
  int ts = (t * TT0) / TT1, te = ((t + 1) * TT0 + TT1 - 1) / TT1;
  int fs = (f * FF0) / FF1, fe = ((f + 1) * FF0 + FF1 - 1) / FF1;
  const float* src = h0 + ((long)(b * CHC + c)) * TT0 * FF0;
  float s = 0.f;
  for (int tt = ts; tt < te; ++tt)
    for (int ff = fs; ff < fe; ++ff) s += src[tt * FF0 + ff];
  float cnt = (float)((te - ts) * (fe - fs));
  xdp[i] = h1[(((long)(b * CHC + c)) * TT1 + t) * FF1 + f] + s / cnt;
}

// LayerNorm over channels (64) per (b,t,f)
__global__ void k_ln_c(const float* __restrict__ x, float* __restrict__ y,
                       const float* g, const float* bt) {
  int i = blockIdx.x * 256 + threadIdx.x;
  if (i >= PATT) return;
  int f = i % FDD; int t = (i / FDD) % TT1; int b = i / (FDD * TT1);
  long base = (long)b * PERB1 + (long)t * FDD + f;
  float m = 0.f;
  for (int c = 0; c < 64; ++c) m += x[base + c * TF1];
  m *= (1.f / 64.f);
  float v = 0.f;
  for (int c = 0; c < 64; ++c) { float d = x[base + c * TF1] - m; v += d * d; }
  v *= (1.f / 64.f);
  float inv = rsqrtf(v + EPSV);
  for (int c = 0; c < 64; ++c)
    y[base + c * TF1] = (x[base + c * TF1] - m) * inv * g[c] + bt[c];
}

// path1 unfold: scrambled flat view of [B,C,T,F] -> [498,64,31], windows of 8
__global__ void k_unfold1(const float* __restrict__ xn, bf16* __restrict__ u) {
  long i = (long)blockIdx.x * 256 + threadIdx.x;
  if (i >= (long)MU1 * 512) return;
  int col = (int)(i & 511); long row = i >> 9;
  int w = (int)(row % NW1); long n = row / NW1;
  int cc = col >> 3, kk = col & 7;
  u[i] = tobf(xn[n * 1984 + cc * 31 + w + kk]);
}

// path2 unfold: sequences over time, n2=b*31+f
__global__ void k_unfold2(const float* __restrict__ xn, bf16* __restrict__ u) {
  long i = (long)blockIdx.x * 256 + threadIdx.x;
  if (i >= (long)MU2 * 512) return;
  int col = (int)(i & 511); long row = i >> 9;
  int w = (int)(row % NW2); int n2 = (int)(row / NW2);
  int b = n2 / FDD, f = n2 % FDD;
  int cc = col >> 3, kk = col & 7;
  u[i] = tobf(xn[(((long)(b * CHC + cc)) * TT1 + (w + kk)) * FDD + f]);
}

// Batched bidirectional LSTM scan: one wave per (sequence, direction).
// Gates precomputed (input proj + biases); whh cached in LDS.
__global__ void k_lstm(const float* __restrict__ Gf, const float* __restrict__ Gr,
                       const float* __restrict__ whhf, const float* __restrict__ whhr,
                       float* __restrict__ r, int N, int NW) {
  __shared__ float swhh[2][4096];
  __shared__ float hsh[8][32];
  int tid = threadIdx.x;
  for (int i = tid; i < 4096; i += 256) { swhh[0][i] = whhf[i]; swhh[1][i] = whhr[i]; }
  __syncthreads();
  int wv = tid >> 5, lane = tid & 31;
  int gw = blockIdx.x * 8 + wv;
  bool active = gw < 2 * N;
  int dir = active ? gw / N : 0;
  int n   = active ? gw % N : 0;
  const float* G  = dir ? Gr : Gf;
  const float* wh = swhh[dir];
  float h = 0.f, c = 0.f;
  for (int s = 0; s < NW; ++s) {
    hsh[wv][lane] = h;
    __syncthreads();
    int w = dir ? (NW - 1 - s) : s;
    const float* grow = G + ((long)n * NW + w) * 128;
    float gi = grow[lane], gf = grow[32 + lane], gg = grow[64 + lane], go = grow[96 + lane];
    const float* hv = hsh[wv];
    #pragma unroll 4
    for (int l = 0; l < 32; ++l) {
      float hl = hv[l];
      gi += hl * wh[lane * 32 + l];
      gf += hl * wh[(32 + lane) * 32 + l];
      gg += hl * wh[(64 + lane) * 32 + l];
      go += hl * wh[(96 + lane) * 32 + l];
    }
    c = sigm(gf) * c + sigm(gi) * tanhf(gg);
    h = sigm(go) * tanhf(c);
    if (active) r[((long)n * NW + w) * 64 + dir * 32 + lane] = h;
    __syncthreads();
  }
}

// ConvTranspose1d(stride1, full pad) + residual add into x
__global__ void k_convt(const float* __restrict__ r, const float* __restrict__ tw,
                        const float* __restrict__ tb, float* __restrict__ x,
                        int mode, int NW) {
  long i = (long)blockIdx.x * 256 + threadIdx.x;
  if (i >= (long)BB * PERB1) return;
  int f = (int)(i % FDD); int t = (int)((i / FDD) % TT1);
  int c = (int)((i / TF1) & 63); int b = (int)(i / PERB1);
  int n, l;
  if (mode == 0) { n = b * TT1 + t; l = f; } else { n = b * FDD + f; l = t; }
  float s = tb[c];
  int lo = l - (NW - 1); if (lo < 0) lo = 0;
  int hi = l < 7 ? l : 7;
  for (int kk = lo; kk <= hi; ++kk) {
    const float* rr = r + ((long)n * NW + (l - kk)) * 64;
    const float* ww = tw + c * 8 + kk;   // tw[j][c][kk]
    float a = 0.f;
    #pragma unroll 8
    for (int j = 0; j < 64; ++j) a += rr[j] * ww[j * 512];
    s += a;
  }
  x[i] += s;
}

// pack [B,C,T,F] -> bf16 [p=(b,t,f)][c]
__global__ void k_pack_dp(const float* __restrict__ x, bf16* __restrict__ d) {
  long i = (long)blockIdx.x * 256 + threadIdx.x;
  if (i >= (long)PATT * 64) return;
  int c = (int)(i & 63); long p = i >> 6;
  int f = (int)(p % FDD); int t = (int)((p / FDD) % TT1); int b = (int)(p / TF1);
  d[i] = tobf(x[(((long)(b * CHC + c)) * TT1 + t) * FDD + f]);
}

// Q/K post: +PReLU, LN over (d,f)=124, write bf16 [hb][t][128] (pad zeroed)
__global__ void k_qk_post(const float* __restrict__ Y, const float* a,
                          const float* g, const float* beta, bf16* __restrict__ out) {
  int grp = blockIdx.x * 8 + (threadIdx.x >> 5);
  int lane = threadIdx.x & 31;
  if (grp >= NHH * BB * TT1) return;
  int t = grp % TT1; int b = (grp / TT1) & 1; int h = grp / (BB * TT1);
  long pbase = ((long)b * TT1 + t) * FDD;
  float ah = a[h], s1 = 0.f, s2 = 0.f;
  for (int i = lane; i < 124; i += 32) {
    int d = i / 31, f = i % 31;
    float v = Y[(pbase + f) * 16 + h * 4 + d];
    v = v > 0.f ? v : ah * v;
    s1 += v; s2 += v * v;
  }
  s1 = wred(s1); s2 = wred(s2);
  float m = s1 / 124.f, var = s2 / 124.f - m * m, inv = rsqrtf(var + EPSV);
  bf16* o = out + ((long)(h * BB + b) * TT1 + t) * 128;
  for (int i = lane; i < 128; i += 32) {
    if (i < 124) {
      int d = i / 31, f = i % 31;
      float v = Y[(pbase + f) * 16 + h * 4 + d];
      v = v > 0.f ? v : ah * v;
      o[i] = tobf((v - m) * inv * g[h * 124 + i] + beta[h * 124 + i]);
    } else o[i] = tobf(0.f);
  }
}

// V post: +PReLU, LN over (cv,f)=496, write Vw[hb][j=cv*31+f][t] bf16 (t pad pre-zeroed)
__global__ void k_v_post(const float* __restrict__ Y, const float* a,
                         const float* g, const float* beta, bf16* __restrict__ Vw) {
  int grp = blockIdx.x * 8 + (threadIdx.x >> 5);
  int lane = threadIdx.x & 31;
  if (grp >= NHH * BB * TT1) return;
  int t = grp % TT1; int b = (grp / TT1) & 1; int h = grp / (BB * TT1);
  long pbase = ((long)b * TT1 + t) * FDD;
  float ah = a[h], s1 = 0.f, s2 = 0.f;
  for (int i = lane; i < 496; i += 32) {
    int cv = i / 31, f = i % 31;
    float v = Y[(pbase + f) * 64 + h * 16 + cv];
    v = v > 0.f ? v : ah * v;
    s1 += v; s2 += v * v;
  }
  s1 = wred(s1); s2 = wred(s2);
  float m = s1 / 496.f, var = s2 / 496.f - m * m, inv = rsqrtf(var + EPSV);
  bf16* o = Vw + (long)(h * BB + b) * 496 * 256;
  for (int i = lane; i < 496; i += 32) {
    int cv = i / 31, f = i % 31;
    float v = Y[(pbase + f) * 64 + h * 16 + cv];
    v = v > 0.f ? v : ah * v;
    o[(long)i * 256 + t] = tobf((v - m) * inv * g[h * 496 + i] + beta[h * 496 + i]);
  }
}

// softmax over 249 cols; write bf16 P (cols 249..255 pre-zeroed)
__global__ void k_softmax(const float* __restrict__ S, bf16* __restrict__ P) {
  int row = blockIdx.x * 8 + (threadIdx.x >> 5);
  int lane = threadIdx.x & 31;
  if (row >= 8 * TT1) return;
  const float* s = S + (long)row * 256;
  float scale = rsqrtf(124.f);
  float mx = -3.4e38f;
  for (int i = lane; i < TT1; i += 32) mx = fmaxf(mx, s[i] * scale);
  mx = wredmax(mx);
  float sum = 0.f;
  for (int i = lane; i < TT1; i += 32) sum += __expf(s[i] * scale - mx);
  sum = wred(sum);
  float inv = 1.f / sum;
  bf16* p = P + (long)row * 256;
  for (int i = lane; i < TT1; i += 32) p[i] = tobf(__expf(s[i] * scale - mx) * inv);
}

// ctx[hb][t][j] -> bf16 [p=(b,t,f)][c=h*16+cv]
__global__ void k_attn_pack(const float* __restrict__ ctx, bf16* __restrict__ Ap) {
  long i = (long)blockIdx.x * 256 + threadIdx.x;
  if (i >= (long)PATT * 64) return;
  int c = (int)(i & 63); long p = i >> 6;
  int f = (int)(p % FDD); int t = (int)((p / FDD) % TT1); int b = (int)(p / TF1);
  int h = c >> 4, cv = c & 15;
  Ap[i] = tobf(ctx[((long)(h * BB + b) * TT1 + t) * 496 + cv * 31 + f]);
}

// proj post: PReLU(scalar), LN over (c,f)=1984 per (b,t), + residual x_dp -> hdp
__global__ void k_proj_post(const float* __restrict__ Yp, const float* pa,
                            const float* g, const float* beta,
                            const float* __restrict__ xdp, float* __restrict__ hdp) {
  int grp = blockIdx.x * 8 + (threadIdx.x >> 5);
  int lane = threadIdx.x & 31;
  if (grp >= BB * TT1) return;
  int b = grp / TT1, t = grp % TT1;
  float a = pa[0], s1 = 0.f, s2 = 0.f;
  for (int i = lane; i < 1984; i += 32) {
    int c = i / 31, f = i % 31;
    float v = Yp[(((long)b * TT1 + t) * FDD + f) * 64 + c];
    v = v > 0.f ? v : a * v;
    s1 += v; s2 += v * v;
  }
  s1 = wred(s1); s2 = wred(s2);
  float m = s1 / 1984.f, var = s2 / 1984.f - m * m, inv = rsqrtf(var + EPSV);
  for (int i = lane; i < 1984; i += 32) {
    int c = i / 31, f = i % 31;
    float v = Yp[(((long)b * TT1 + t) * FDD + f) * 64 + c];
    v = v > 0.f ? v : a * v;
    long idx = (((long)(b * CHC + c)) * TT1 + t) * FDD + f;
    hdp[idx] = (v - m) * inv * g[i] + beta[i] + xdp[idx];
  }
}

// GroupNorm(1) stats, stage 1: partial (sum,sumsq) of src*w[c]+b[c] per sample
__global__ void k_stat1(const float* __restrict__ src, const float* w, const float* bia,
                        long perS, long HW, float* __restrict__ part) {
  int b = blockIdx.y;
  const float* s = src + (long)b * perS;
  float s1 = 0.f, s2 = 0.f;
  for (long i = (long)blockIdx.x * 256 + threadIdx.x; i < perS; i += 256L * 256) {
    int c = (int)((i / HW) & 63);
    float v = s[i] * w[c] + bia[c];
    s1 += v; s2 += v * v;
  }
  __shared__ float sh1[256], sh2[256];
  sh1[threadIdx.x] = s1; sh2[threadIdx.x] = s2;
  __syncthreads();
  for (int st = 128; st; st >>= 1) {
    if (threadIdx.x < st) { sh1[threadIdx.x] += sh1[threadIdx.x + st]; sh2[threadIdx.x] += sh2[threadIdx.x + st]; }
    __syncthreads();
  }
  if (threadIdx.x == 0) {
    part[(b * 256 + blockIdx.x) * 2]     = sh1[0];
    part[(b * 256 + blockIdx.x) * 2 + 1] = sh2[0];
  }
}
__global__ void k_stat2(const float* __restrict__ part, float* __restrict__ stats,
                        int slot, float invN) {
  __shared__ float s1[256], s2[256];
  int b = blockIdx.x, t = threadIdx.x;
  s1[t] = part[(b * 256 + t) * 2]; s2[t] = part[(b * 256 + t) * 2 + 1];
  __syncthreads();
  for (int st = 128; st; st >>= 1) {
    if (t < st) { s1[t] += s1[t + st]; s2[t] += s2[t + st]; }
    __syncthreads();
  }
  if (t == 0) {
    float m = s1[0] * invN, var = s2[0] * invN - m * m;
    stats[(slot * 2 + b) * 2] = m;
    stats[(slot * 2 + b) * 2 + 1] = rsqrtf(var + EPSV);
  }
}

// o1 = sigmoid(gn1(src*w1+b1)); o3 = gn3(src*w3+b3)
__global__ void k_w1w3(const float* __restrict__ src, long perC,
                       const float* __restrict__ stats, int slotA, int slotB,
                       const float* w1, const float* b1, const float* g1, const float* be1,
                       const float* w3, const float* b3, const float* g3, const float* be3,
                       float* __restrict__ o1, float* __restrict__ o3, long total) {
  long i = (long)blockIdx.x * 256 + threadIdx.x;
  if (i >= total) return;
  int c = (int)((i / perC) & 63); int b = (int)(i / (perC * 64));
  float x = src[i];
  float mA = stats[(slotA * 2 + b) * 2], iA = stats[(slotA * 2 + b) * 2 + 1];
  float v1 = ((x * w1[c] + b1[c]) - mA) * iA * g1[c] + be1[c];
  o1[i] = sigm(v1);
  float mB = stats[(slotB * 2 + b) * 2], iB = stats[(slotB * 2 + b) * 2 + 1];
  o3[i] = ((x * w3[c] + b3[c]) - mB) * iB * g3[c] + be3[c];
}

// out = resize(w1n)*gn2(m) + resize(w3n)  (w1n/w3n have dims [b][c][249][31])
__global__ void k_nA(const float* __restrict__ m, long mT, long mF,
                     const float* __restrict__ w1n, const float* __restrict__ w3n,
                     const float* __restrict__ stats, int slot,
                     const float* w2, const float* b2, const float* g2, const float* be2,
                     float* __restrict__ out, long total) {
  long i = (long)blockIdx.x * 256 + threadIdx.x;
  if (i >= total) return;
  long f = i % mF; long t = (i / mF) % mT;
  int c = (int)((i / (mF * mT)) & 63); int b = (int)(i / (mF * mT * 64));
  long ti = (t * 249L) / mT, fi = (f * 31L) / mF;
  float mm = stats[(slot * 2 + b) * 2], iv = stats[(slot * 2 + b) * 2 + 1];
  float x2 = ((m[i] * w2[c] + b2[c]) - mm) * iv * g2[c] + be2[c];
  long ni = (((long)(b * 64 + c)) * 249 + ti) * 31 + fi;
  out[i] = w1n[ni] * x2 + w3n[ni];
}

// rec = resize(w1n2)*gn2(nA0) + resize(w3n2) + h0, packed bf16 [b][tf][c] for up-GEMM
__global__ void k_rec_pack(const float* __restrict__ nA0, const float* __restrict__ w1n2,
                           const float* __restrict__ w3n2, const float* __restrict__ h0,
                           const float* __restrict__ stats, int slot,
                           const float* w2, const float* b2, const float* g2, const float* be2,
                           bf16* __restrict__ recT) {
  long i = (long)blockIdx.x * 256 + threadIdx.x;
  if (i >= 8320000L) return;
  long f = i % FF0; long t = (i / FF0) % TT0;
  int c = (int)((i / TF0) & 63); int b = (int)(i / (TF0 * 64));
  long ti = (t * 249L) / 500, fi = (f * 64L) / 130;
  long ni = (((long)(b * 64 + c)) * 249 + ti) * 64 + fi;
  float mm = stats[(slot * 2 + b) * 2], iv = stats[(slot * 2 + b) * 2 + 1];
  float x2 = ((nA0[i] * w2[c] + b2[c]) - mm) * iv * g2[c] + be2[c];
  float v = w1n2[ni] * x2 + w3n2[ni] + h0[i];
  recT[((long)b * TF0 + t * FF0 + f) * 64 + c] = tobf(v);
}

// ===================== host side =====================
#define GRID1(n) dim3((unsigned)(((n) + 255) / 256))

static void gemm(hipStream_t s, const void* A, long aB, int lda,
                 const void* W, long wB, int ldw,
                 float* C, long cB, long rs, long cs,
                 const float* bias, int M, int N, int K, int batch) {
  int tm = (M + 15) / 16, tn = (N + 15) / 16;
  dim3 g((unsigned)((tm * tn + 3) / 4), (unsigned)batch);
  k_gemm<<<g, 128, 0, s>>>((const bf16*)A, aB, lda, (const bf16*)W, wB, ldw,
                           C, cB, rs, cs, bias, M, N, K);
}

// workspace byte offsets (arena; phase region R is reused)
#define O_H0   0L
#define O_H1   33280000L
#define O_XDP  41439232L
#define O_HDP  45391360L
#define O_WB   49343488L
#define O_R    50392064L
#define WB_DOWNW 0L
#define WB_UPW   32768L
#define WB_WIH1F 65536L
#define WB_WIH1R 196608L
#define WB_WIH2F 327680L
#define WB_WIH2R 458752L
#define WB_QW    589824L
#define WB_KW    591872L
#define WB_VW    593920L
#define WB_PW    602112L
#define WB_B1F   610304L
#define WB_B1R   610816L
#define WB_B2F   611328L
#define WB_B2R   611840L
#define WB_STATS 612352L
#define WB_PART  612608L
#define R_XT   0L
#define R_XN   0L
#define R_U    4194304L
#define R_GF   20971520L
#define R_GR   29360128L
#define R_RQ   37748736L
#define R_DPA  0L
#define R_YQ   2097152L
#define R_YK   3145728L
#define R_YV   4194304L
#define R_QF   8388608L
#define R_KF   9437184L
#define R_VWB  10485760L
#define R_S    12582912L
#define R_PBF  14680064L
#define R_CTX  15728640L
#define R_AP   19922944L
#define R_YP   22020096L
#define R_W1N  0L
#define R_W3N  4194304L
#define R_NA0  8388608L
#define R_NA1  41943040L
#define R_W1N2 50331648L
#define R_W3N2 58720256L
#define R_RECT 67108864L

static void run_stat(hipStream_t s, const float* src, const float* w, const float* b,
                     long perS, long HW, float* part, float* stats, int slot) {
  k_stat1<<<dim3(256, BB), 256, 0, s>>>(src, w, b, perS, HW, part);
  k_stat2<<<BB, 256, 0, s>>>(part, stats, slot, 1.f / (float)perS);
}

extern "C" void kernel_launch(void* const* d_in, const int* in_sizes, int n_in,
                              void* d_out, int out_size, void* d_ws, size_t ws_size,
                              hipStream_t stream) {
  (void)in_sizes; (void)n_in; (void)out_size; (void)ws_size;
  const float* I[80];
  for (int i = 0; i < n_in && i < 80; ++i) I[i] = (const float*)d_in[i];
  // input index map (setup_inputs() recursive dict insertion order):
  const float* X      = I[0];
  const float* DOWNW  = I[1];  const float* DOWNB = I[2];
  const float* B0W = I[3], *B0B = I[4], *B0G = I[5], *B0BE = I[6], *B0M = I[7], *B0V = I[8];
  const float* LN1G = I[15], *LN1B = I[16], *LN2G = I[17], *LN2B = I[18];
  const float* WIH1F = I[19], *WHH1F = I[20], *BIH1F = I[21], *BHH1F = I[22];
  const float* WIH1R = I[23], *WHH1R = I[24], *BIH1R = I[25], *BHH1R = I[26];
  const float* WIH2F = I[27], *WHH2F = I[28], *BIH2F = I[29], *BHH2F = I[30];
  const float* WIH2R = I[31], *WHH2R = I[32], *BIH2R = I[33], *BHH2R = I[34];
  const float* TC1W = I[35], *TC1B = I[36], *TC2W = I[37], *TC2B = I[38];
  const float* QW = I[39], *QB = I[40], *QA = I[41], *QG = I[42], *QBE = I[43];
  const float* KW = I[44], *KB = I[45], *KA = I[46], *KG = I[47], *KBE = I[48];
  const float* VW = I[49], *VB = I[50], *VA = I[51], *VG = I[52], *VBE = I[53];
  const float* PW = I[54], *PB = I[55], *PA = I[56], *PG = I[57], *PBE = I[58];
  const float* W1W = I[59], *W1B = I[60], *W1G = I[61], *W1BE = I[62];
  const float* W2W = I[63], *W2B = I[64], *W2G = I[65], *W2BE = I[66];
  const float* W3W = I[67], *W3B = I[68], *W3G = I[69], *W3BE = I[70];
  const float* UPW = I[71], *UPB = I[72];

  char* ws = (char*)d_ws;
  float* h0  = (float*)(ws + O_H0);
  float* h1  = (float*)(ws + O_H1);
  float* xdp = (float*)(ws + O_XDP);
  float* hdp = (float*)(ws + O_HDP);
  char*  wb  = ws + O_WB;
  char*  R   = ws + O_R;
  float* STATS = (float*)(wb + WB_STATS);
  float* PART  = (float*)(wb + WB_PART);
  float* OUT = (float*)d_out;

  // ---- weight conversions (bf16) + combined LSTM biases ----
  k_cvt<<<GRID1(16384), 256, 0, stream>>>(DOWNW, (bf16*)(wb + WB_DOWNW), 16384);
  k_cvt<<<GRID1(16384), 256, 0, stream>>>(UPW,   (bf16*)(wb + WB_UPW),   16384);
  k_cvt<<<GRID1(65536), 256, 0, stream>>>(WIH1F, (bf16*)(wb + WB_WIH1F), 65536);
  k_cvt<<<GRID1(65536), 256, 0, stream>>>(WIH1R, (bf16*)(wb + WB_WIH1R), 65536);
  k_cvt<<<GRID1(65536), 256, 0, stream>>>(WIH2F, (bf16*)(wb + WB_WIH2F), 65536);
  k_cvt<<<GRID1(65536), 256, 0, stream>>>(WIH2R, (bf16*)(wb + WB_WIH2R), 65536);
  k_cvt<<<GRID1(1024), 256, 0, stream>>>(QW, (bf16*)(wb + WB_QW), 1024);
  k_cvt<<<GRID1(1024), 256, 0, stream>>>(KW, (bf16*)(wb + WB_KW), 1024);
  k_cvt<<<GRID1(4096), 256, 0, stream>>>(VW, (bf16*)(wb + WB_VW), 4096);
  k_cvt<<<GRID1(4096), 256, 0, stream>>>(PW, (bf16*)(wb + WB_PW), 4096);
  k_addb<<<1, 256, 0, stream>>>(BIH1F, BHH1F, (float*)(wb + WB_B1F), 128);
  k_addb<<<1, 256, 0, stream>>>(BIH1R, BHH1R, (float*)(wb + WB_B1R), 128);
  k_addb<<<1, 256, 0, stream>>>(BIH2F, BHH2F, (float*)(wb + WB_B2F), 128);
  k_addb<<<1, 256, 0, stream>>>(BIH2R, BHH2R, (float*)(wb + WB_B2R), 128);

  // ---- down conv1x1 (WMMA GEMM), dwconv+bn+relu, pool+crop ----
  k_pack_x<<<GRID1(33280000L), 256, 0, stream>>>(X, (bf16*)(R + R_XT));
  gemm(stream, R + R_XT, TF0 * 256, 256, wb + WB_DOWNW, 0, 256,
       h0, 64 * TF0, 1, TF0, DOWNB, 65000, 64, 256, BB);
  k_dw1<<<GRID1(2039808L), 256, 0, stream>>>(h0, h1, B0W, B0B, B0G, B0BE, B0M, B0V);
  k_pool_add<<<GRID1(988032L), 256, 0, stream>>>(h0, h1, xdp);

  // ---- dual path 1 (sequences along the scrambled flat view) ----
  float* XN = (float*)(R + R_XN);
  bf16*  U  = (bf16*)(R + R_U);
  float* GF = (float*)(R + R_GF);
  float* GR = (float*)(R + R_GR);
  float* RQ = (float*)(R + R_RQ);
  k_ln_c<<<GRID1(PATT), 256, 0, stream>>>(xdp, XN, LN1G, LN1B);
  k_unfold1<<<GRID1((long)MU1 * 512), 256, 0, stream>>>(XN, U);
  gemm(stream, U, 0, 512, wb + WB_WIH1F, 0, 512, GF, 0, 128, 1,
       (float*)(wb + WB_B1F), MU1, 128, 512, 1);
  gemm(stream, U, 0, 512, wb + WB_WIH1R, 0, 512, GR, 0, 128, 1,
       (float*)(wb + WB_B1R), MU1, 128, 512, 1);
  k_lstm<<<dim3((2 * NSEQ1 + 7) / 8), 256, 0, stream>>>(GF, GR, WHH1F, WHH1R, RQ, NSEQ1, NW1);
  k_convt<<<GRID1(988032L), 256, 0, stream>>>(RQ, TC1W, TC1B, xdp, 0, NW1);

  // ---- dual path 2 (sequences along time) ----
  k_ln_c<<<GRID1(PATT), 256, 0, stream>>>(xdp, XN, LN2G, LN2B);
  k_unfold2<<<GRID1((long)MU2 * 512), 256, 0, stream>>>(XN, U);
  gemm(stream, U, 0, 512, wb + WB_WIH2F, 0, 512, GF, 0, 128, 1,
       (float*)(wb + WB_B2F), MU2, 128, 512, 1);
  gemm(stream, U, 0, 512, wb + WB_WIH2R, 0, 512, GR, 0, 128, 1,
       (float*)(wb + WB_B2R), MU2, 128, 512, 1);
  k_lstm<<<dim3((2 * NSEQ2 + 7) / 8), 256, 0, stream>>>(GF, GR, WHH2F, WHH2R, RQ, NSEQ2, NW2);
  k_convt<<<GRID1(988032L), 256, 0, stream>>>(RQ, TC2W, TC2B, xdp, 1, NW2);

  // ---- TF attention ----
  bf16* DPA = (bf16*)(R + R_DPA);
  float* YQ = (float*)(R + R_YQ);
  float* YK = (float*)(R + R_YK);
  float* YV = (float*)(R + R_YV);
  bf16* QF = (bf16*)(R + R_QF);
  bf16* KF = (bf16*)(R + R_KF);
  bf16* VWB = (bf16*)(R + R_VWB);
  float* SS = (float*)(R + R_S);
  bf16* PBF = (bf16*)(R + R_PBF);
  float* CTX = (float*)(R + R_CTX);
  bf16* AP = (bf16*)(R + R_AP);
  float* YP = (float*)(R + R_YP);

  k_pack_dp<<<GRID1((long)PATT * 64), 256, 0, stream>>>(xdp, DPA);
  gemm(stream, DPA, 0, 64, wb + WB_QW, 0, 64, YQ, 0, 16, 1, QB, PATT, 16, 64, 1);
  gemm(stream, DPA, 0, 64, wb + WB_KW, 0, 64, YK, 0, 16, 1, KB, PATT, 16, 64, 1);
  gemm(stream, DPA, 0, 64, wb + WB_VW, 0, 64, YV, 0, 64, 1, VB, PATT, 64, 64, 1);
  k_zero<<<GRID1(507904L), 256, 0, stream>>>((float*)VWB, 507904L);   // bf16 pad (t>=249)
  k_qk_post<<<dim3(249), 256, 0, stream>>>(YQ, QA, QG, QBE, QF);
  k_qk_post<<<dim3(249), 256, 0, stream>>>(YK, KA, KG, KBE, KF);
  k_v_post<<<dim3(249), 256, 0, stream>>>(YV, VA, VG, VBE, VWB);
  gemm(stream, QF, 249L * 128, 128, KF, 249L * 128, 128,
       SS, 249L * 256, 256, 1, nullptr, 249, 249, 128, 8);
  k_zero<<<GRID1(254976L), 256, 0, stream>>>((float*)PBF, 254976L);   // bf16 pad (s>=249)
  k_softmax<<<dim3(249), 256, 0, stream>>>(SS, PBF);
  gemm(stream, PBF, 249L * 256, 256, VWB, 496L * 256, 256,
       CTX, 249L * 496, 496, 1, nullptr, 249, 496, 256, 8);
  k_attn_pack<<<GRID1((long)PATT * 64), 256, 0, stream>>>(CTX, AP);
  gemm(stream, AP, 0, 64, wb + WB_PW, 0, 64, YP, 0, 64, 1, PB, PATT, 64, 64, 1);
  k_proj_post<<<dim3(63), 256, 0, stream>>>(YP, PA, PG, PBE, xdp, hdp);

  // ---- attention reconstruction ----
  float* W1N  = (float*)(R + R_W1N);
  float* W3N  = (float*)(R + R_W3N);
  float* NA0  = (float*)(R + R_NA0);
  float* NA1  = (float*)(R + R_NA1);
  float* W1N2 = (float*)(R + R_W1N2);
  float* W3N2 = (float*)(R + R_W3N2);
  bf16*  RECT = (bf16*)(R + R_RECT);

  run_stat(stream, hdp, W1W, W1B, 494016L, 7719L, PART, STATS, 0);
  run_stat(stream, hdp, W3W, W3B, 494016L, 7719L, PART, STATS, 1);
  run_stat(stream, h0,  W2W, W2B, 4160000L, 65000L, PART, STATS, 2);
  run_stat(stream, h1,  W2W, W2B, 1019904L, 15936L, PART, STATS, 3);
  k_w1w3<<<GRID1(988032L), 256, 0, stream>>>(hdp, 7719L, STATS, 0, 1,
      W1W, W1B, W1G, W1BE, W3W, W3B, W3G, W3BE, W1N, W3N, 988032L);
  k_nA<<<GRID1(8320000L), 256, 0, stream>>>(h0, 500, 130, W1N, W3N, STATS, 2,
      W2W, W2B, W2G, W2BE, NA0, 8320000L);
  k_nA<<<GRID1(2039808L), 256, 0, stream>>>(h1, 249, 64, W1N, W3N, STATS, 3,
      W2W, W2B, W2G, W2BE, NA1, 2039808L);
  run_stat(stream, NA1, W1W, W1B, 1019904L, 15936L, PART, STATS, 4);
  run_stat(stream, NA1, W3W, W3B, 1019904L, 15936L, PART, STATS, 5);
  run_stat(stream, NA0, W2W, W2B, 4160000L, 65000L, PART, STATS, 6);
  k_w1w3<<<GRID1(2039808L), 256, 0, stream>>>(NA1, 15936L, STATS, 4, 5,
      W1W, W1B, W1G, W1BE, W3W, W3B, W3G, W3BE, W1N2, W3N2, 2039808L);
  k_rec_pack<<<GRID1(8320000L), 256, 0, stream>>>(NA0, W1N2, W3N2, h0, STATS, 6,
      W2W, W2B, W2G, W2BE, RECT);

  // ---- up conv1x1 (WMMA GEMM) + residual ----
  gemm(stream, RECT, TF0 * 64, 64, wb + WB_UPW, 0, 64,
       OUT, 256L * TF0, 1, TF0, UPB, 65000, 256, 64, BB);
  k_addres<<<GRID1(33280000L), 256, 0, stream>>>(OUT, X, 33280000L);
}
